// MinGRU_Layers_17128329576474
// MI455X (gfx1250) — matrix operand order repaired and checked by reading, not compiled, not run on hardware
//
#include <hip/hip_runtime.h>
#include <hip/hip_bf16.h>
#include <math.h>
#include <stdint.h>

// ---------------------------------------------------------------------------
// Types matching the gfx1250 WMMA builtin signatures (probe-verified).
// ---------------------------------------------------------------------------
typedef __bf16 bf16_t;
typedef __attribute__((ext_vector_type(16))) __bf16 v16bf;
typedef __attribute__((ext_vector_type(8)))  __bf16 v8bf;
typedef __attribute__((ext_vector_type(8)))  float  v8f;

union Frag16 { v16bf v; v8bf h[2]; };

// CDNA5 async global->LDS path (ASYNCcnt), guarded so absence falls back.
#if defined(__HIP_DEVICE_COMPILE__)
#if __has_builtin(__builtin_amdgcn_global_load_async_to_lds_b128) && \
    __has_builtin(__builtin_amdgcn_s_wait_asynccnt)
#define HAVE_ASYNC_LDS 1
#endif
#endif

#ifdef HAVE_ASYNC_LDS
// Builtin signature (from hipcc diagnostic): (int4 AS1*, int4 AS3*, Ii, Ii)
typedef int b128i __attribute__((vector_size(16)));
typedef __attribute__((address_space(1))) b128i* as1_b128_p;
typedef __attribute__((address_space(3))) b128i* as3_b128_p;
// LDS generic address: low 32 bits are the LDS offset (ISA 10.2 aperture),
// so integer truncation is the portable generic->AS3 conversion.
#define ASYNC_CP16(dst_lds, src_glb)                                      \
  __builtin_amdgcn_global_load_async_to_lds_b128(                         \
      (as1_b128_p)(uintptr_t)(src_glb),                                   \
      (as3_b128_p)(uint32_t)(uintptr_t)(dst_lds), 0, 0)
#endif

// ---------------------------------------------------------------------------
// Problem constants
// ---------------------------------------------------------------------------
constexpr int D_  = 1024;
constexpr int V_  = 50257;
constexpr int VP_ = 50304;          // V padded to 393*128 for tile alignment
constexpr int B_  = 2;
constexpr int S1_ = 2049;
constexpr int S_  = 2048;
constexpr int T_  = B_ * S_;        // 4096 tokens
constexpr int C_  = 32;             // scan chunks
constexpr int L_  = S_ / C_;        // 64 steps per chunk

// ---------------------------------------------------------------------------
// Small math helpers
// ---------------------------------------------------------------------------
__device__ __forceinline__ float softplus_f(float x) {
  return log1pf(expf(-fabsf(x))) + fmaxf(x, 0.0f);
}
// logaddexp(a,b); safe when either input is -inf (but not both)
__device__ __forceinline__ float logaddexp_f(float a, float b) {
  const float m = fmaxf(a, b);
  return m + log1pf(expf(fminf(a, b) - m));
}

// ---------------------------------------------------------------------------
// fp32 -> bf16 converts
// ---------------------------------------------------------------------------
__global__ __launch_bounds__(256) void cvt_bf16_k(const float* __restrict__ s,
                                                  bf16_t* __restrict__ d, int n) {
  int i = (blockIdx.x * 256 + threadIdx.x) * 4;
  if (i + 3 < n) {
    float4 f = *(const float4*)(s + i);
    d[i + 0] = (bf16_t)f.x; d[i + 1] = (bf16_t)f.y;
    d[i + 2] = (bf16_t)f.z; d[i + 3] = (bf16_t)f.w;
  }
}

// W_logits (D x V) fp32 -> bf16 padded to (D x VP), zero fill in pad columns.
__global__ __launch_bounds__(256) void cvt_wl_pad_k(const float* __restrict__ s,
                                                    bf16_t* __restrict__ d) {
  int i = blockIdx.x * 256 + threadIdx.x;
  if (i >= D_ * VP_) return;
  int r = i / VP_, c = i - r * VP_;
  d[i] = (c < V_) ? (bf16_t)s[(size_t)r * V_ + c] : (bf16_t)0.0f;
}

// ---------------------------------------------------------------------------
// Fused embedding lookup + RMSNorm (gamma1) -> fp32 + bf16
// ---------------------------------------------------------------------------
__global__ __launch_bounds__(256) void embed_rmsnorm_k(
    const int* __restrict__ x, const float* __restrict__ emb,
    const float* __restrict__ gamma, float* __restrict__ h0f,
    bf16_t* __restrict__ h0b) {
  const int t   = blockIdx.x;
  const int b   = t / S_;
  const int s   = t - b * S_;
  const int id  = x[b * S1_ + s];
  const int tid = threadIdx.x;

  float4 f = ((const float4*)(emb + (size_t)id * D_))[tid];
  float ss = f.x * f.x + f.y * f.y + f.z * f.z + f.w * f.w;

  __shared__ float red[256];
  red[tid] = ss; __syncthreads();
  for (int o = 128; o > 0; o >>= 1) {
    if (tid < o) red[tid] += red[tid + o];
    __syncthreads();
  }
  const float sc = 32.0f / fmaxf(sqrtf(red[0]), 1e-12f);  // sqrt(1024)=32
  float4 g = ((const float4*)gamma)[tid];

  float o0 = f.x * sc * (g.x + 1.0f);
  float o1 = f.y * sc * (g.y + 1.0f);
  float o2 = f.z * sc * (g.z + 1.0f);
  float o3 = f.w * sc * (g.w + 1.0f);

  float*  pf = h0f + (size_t)t * D_ + tid * 4;
  bf16_t* pb = h0b + (size_t)t * D_ + tid * 4;
  pf[0] = o0; pf[1] = o1; pf[2] = o2; pf[3] = o3;
  pb[0] = (bf16_t)o0; pb[1] = (bf16_t)o1; pb[2] = (bf16_t)o2; pb[3] = (bf16_t)o3;
}

// RMSNorm (gamma2) over fp32 rows -> bf16 only.
__global__ __launch_bounds__(256) void rmsnorm_bf16_k(
    const float* __restrict__ src, const float* __restrict__ gamma,
    bf16_t* __restrict__ outb) {
  const int t = blockIdx.x, tid = threadIdx.x;
  float4 f = ((const float4*)(src + (size_t)t * D_))[tid];
  float ss = f.x * f.x + f.y * f.y + f.z * f.z + f.w * f.w;

  __shared__ float red[256];
  red[tid] = ss; __syncthreads();
  for (int o = 128; o > 0; o >>= 1) {
    if (tid < o) red[tid] += red[tid + o];
    __syncthreads();
  }
  const float sc = 32.0f / fmaxf(sqrtf(red[0]), 1e-12f);
  float4 g = ((const float4*)gamma)[tid];
  bf16_t* pb = outb + (size_t)t * D_ + tid * 4;
  pb[0] = (bf16_t)(f.x * sc * (g.x + 1.0f));
  pb[1] = (bf16_t)(f.y * sc * (g.y + 1.0f));
  pb[2] = (bf16_t)(f.z * sc * (g.z + 1.0f));
  pb[3] = (bf16_t)(f.w * sc * (g.w + 1.0f));
}

// ---------------------------------------------------------------------------
// WMMA bf16 GEMM: C[M,N] = A[M,K] x B[K,N] (+ epilogue)
// 256 thr (8 waves), tile 128x128x32, wave grid 2x4 -> 8 WMMA / k-step.
// Software pipeline with double-buffered LDS: next tile's global loads are
// issued (async-to-LDS when available) before computing the current tile;
// one barrier per iteration.
// ---------------------------------------------------------------------------
constexpr int BM = 128, BN = 128, BK = 32;
constexpr int LDA_S = BK + 8;   // 40 bf16 -> 80B rows, 16B-aligned, conflict-free
constexpr int LDB_S = BK + 8;

enum { EP_PLAIN = 0, EP_GELU = 1, EP_RES = 2, EP_LOGITS = 3 };

template <int EP>
__global__ __launch_bounds__(256) void gemm_wmma_bf16(
    const bf16_t* __restrict__ A, const bf16_t* __restrict__ Bw,
    float* __restrict__ Cf, bf16_t* __restrict__ Cb,
    const float* __restrict__ bias, const float* __restrict__ resid,
    int M, int N, int K, int ldb, int ldc, int Nout) {
  __shared__ alignas(16) bf16_t As[2][BM * LDA_S];
  __shared__ alignas(16) bf16_t Bs[2][BN * LDB_S];

  const int tid    = threadIdx.x;
  const int lane   = tid & 31;
  const int wid    = tid >> 5;
  const int wave_m = wid >> 2;       // 0..1
  const int wave_n = wid & 3;        // 0..3
  const int m0     = blockIdx.y * BM;
  const int n0     = blockIdx.x * BN;

  // Staging roles
  const int a_row = tid >> 1;            // 0..127
  const int a_col = (tid & 1) * 16;      // 0 / 16
  const int b_k   = tid >> 3;            // 0..31
  const int b_n   = (tid & 7) * 16;      // 0..112

  const bf16_t* ga_base = A + (size_t)(m0 + a_row) * K + a_col;
  const bf16_t* gb_base = Bw + (size_t)b_k * ldb + n0 + b_n;
  const size_t  gb_step = (size_t)BK * ldb;

  v8f acc[4][2] = {};

  // ---- prologue: stage tile 0 into buffer 0
  {
#ifdef HAVE_ASYNC_LDS
    ASYNC_CP16(&As[0][a_row * LDA_S + a_col],     ga_base);
    ASYNC_CP16(&As[0][a_row * LDA_S + a_col + 8], ga_base + 8);
#else
    v8bf a0 = *(const v8bf*)ga_base;
    v8bf a1 = *(const v8bf*)(ga_base + 8);
#endif
    Frag16 tb;
    tb.h[0] = *(const v8bf*)gb_base;
    tb.h[1] = *(const v8bf*)(gb_base + 8);
#ifndef HAVE_ASYNC_LDS
    *(v8bf*)&As[0][a_row * LDA_S + a_col]     = a0;
    *(v8bf*)&As[0][a_row * LDA_S + a_col + 8] = a1;
#endif
#pragma unroll
    for (int i = 0; i < 16; ++i) Bs[0][(b_n + i) * LDB_S + b_k] = tb.v[i];
#ifdef HAVE_ASYNC_LDS
    __builtin_amdgcn_s_wait_asynccnt(0);
#endif
  }
  __syncthreads();

  const int ksteps = K / BK;
  for (int kt = 0; kt < ksteps; ++kt) {
    const int cur = kt & 1, nxt = cur ^ 1;
    const bool have_next = (kt + 1) < ksteps;

    // Issue next tile's global loads early (overlap with WMMA below).
    Frag16 nb = {};
#ifndef HAVE_ASYNC_LDS
    v8bf na0 = {}, na1 = {};
#endif
    if (have_next) {
      const bf16_t* ga = ga_base + (size_t)(kt + 1) * BK;
      const bf16_t* gb = gb_base + (size_t)(kt + 1) * gb_step;
#ifdef HAVE_ASYNC_LDS
      ASYNC_CP16(&As[nxt][a_row * LDA_S + a_col],     ga);
      ASYNC_CP16(&As[nxt][a_row * LDA_S + a_col + 8], ga + 8);
#else
      na0 = *(const v8bf*)ga;
      na1 = *(const v8bf*)(ga + 8);
#endif
      nb.h[0] = *(const v8bf*)gb;
      nb.h[1] = *(const v8bf*)(gb + 8);
    }

    // Fragment loads (documented 16-bit A/B VGPR layouts) + 8 WMMAs.
    Frag16 afrag[4], bfrag[2];
    const int am = lane & 15;
    const int kb = (lane < 16) ? 0 : 8;
#pragma unroll
    for (int mi = 0; mi < 4; ++mi) {
      const bf16_t* p = &As[cur][(wave_m * 64 + mi * 16 + am) * LDA_S + kb];
      afrag[mi].h[0] = *(const v8bf*)p;          // K = kb .. kb+7
      afrag[mi].h[1] = *(const v8bf*)(p + 16);   // K = 16+kb .. 16+kb+7
    }
    const int bn   = lane & 15;
    const int koff = (lane >> 4) * 8;
#pragma unroll
    for (int ni = 0; ni < 2; ++ni) {
      const bf16_t* p = &Bs[cur][(wave_n * 32 + ni * 16 + bn) * LDB_S + koff];
      bfrag[ni].h[0] = *(const v8bf*)p;          // K = koff .. koff+7
      bfrag[ni].h[1] = *(const v8bf*)(p + 16);   // K = 16+koff .. 16+koff+7
    }
#pragma unroll
    for (int mi = 0; mi < 4; ++mi)
#pragma unroll
      for (int ni = 0; ni < 2; ++ni)
        acc[mi][ni] = __builtin_amdgcn_wmma_f32_16x16x32_bf16(
            false, afrag[mi].v, false, bfrag[ni].v, (short)0, acc[mi][ni],
            false, false);

    // Drain next tile into the other buffer.
    if (have_next) {
#ifndef HAVE_ASYNC_LDS
      *(v8bf*)&As[nxt][a_row * LDA_S + a_col]     = na0;
      *(v8bf*)&As[nxt][a_row * LDA_S + a_col + 8] = na1;
#endif
#pragma unroll
      for (int i = 0; i < 16; ++i) Bs[nxt][(b_n + i) * LDB_S + b_k] = nb.v[i];
#ifdef HAVE_ASYNC_LDS
      __builtin_amdgcn_s_wait_asynccnt(0);
#endif
    }
    __syncthreads();
  }

  // Epilogue (C/D layout: lanes 0-15 -> N=lane,M=r; lanes 16-31 -> M=r+8)
  const int en = lane & 15;
  const int eh = lane >> 4;
#pragma unroll
  for (int mi = 0; mi < 4; ++mi) {
#pragma unroll
    for (int ni = 0; ni < 2; ++ni) {
#pragma unroll
      for (int r = 0; r < 8; ++r) {
        const int m = m0 + wave_m * 64 + mi * 16 + eh * 8 + r;
        const int n = n0 + wave_n * 32 + ni * 16 + en;
        float v = acc[mi][ni][r];
        if (EP == EP_PLAIN) {
          Cf[(size_t)m * ldc + n] = v;
        } else if (EP == EP_GELU) {
          v += bias[n];
          v = 0.5f * v * (1.0f + erff(v * 0.70710678118654752f));
          Cb[(size_t)m * ldc + n] = (bf16_t)v;
        } else if (EP == EP_RES) {
          v += bias[n] + resid[(size_t)m * ldc + n];
          Cf[(size_t)m * ldc + n] = v;
        } else {  // EP_LOGITS: N-guarded store into packed (ldc = V) output
          if (n < Nout) Cf[(size_t)m * (size_t)ldc + n] = v;
        }
      }
    }
  }
}

// ---------------------------------------------------------------------------
// minGRU scan, chunked parallel form. The step op x -> logaddexp(lc+x, lv)
// composes associatively: (lc2,lv2)o(lc1,lv1) = (lc1+lc2, logaddexp(lc2+lv1,lv2)).
// Stage 1: per (b,chunk,d) compute per-step (lc,lv) (stored) + chunk comp.
// Stage 2: per (b,d) 32-step prefix over chunk summaries -> chunk start states.
// Stage 3: per (b,chunk,d) replay 64 steps from start state, fused epilogue.
// ---------------------------------------------------------------------------
__global__ __launch_bounds__(256) void scan_stage1_k(
    const float* __restrict__ hg, float* __restrict__ lcb,
    float* __restrict__ lvb, float* __restrict__ sum_lc,
    float* __restrict__ sum_lv) {
  const int gi = blockIdx.x * 256 + threadIdx.x;   // over B*C*D
  const int d  = gi % D_;
  const int bc = gi / D_;
  const int c  = bc % C_;
  const int b  = bc / C_;

  float acc_lc = 0.0f, acc_lv = -INFINITY;
  for (int j = 0; j < L_; ++j) {
    const int s    = c * L_ + j;
    const size_t r = (size_t)(b * S_ + s);
    const float hid = hg[r * (2 * D_) + d];
    const float gat = hg[r * (2 * D_) + D_ + d];
    const float lc  = -softplus_f(gat);
    const float lgv = (hid >= 0.0f) ? logf(hid + 0.5f) : -softplus_f(-hid);
    const float lv  = -softplus_f(-gat) + lgv;
    lcb[r * D_ + d] = lc;
    lvb[r * D_ + d] = lv;
    acc_lv = logaddexp_f(lc + acc_lv, lv);
    acc_lc += lc;
  }
  sum_lc[gi] = acc_lc;
  sum_lv[gi] = acc_lv;
}

__global__ __launch_bounds__(256) void scan_stage2_k(
    const float* __restrict__ sum_lc, const float* __restrict__ sum_lv,
    float* __restrict__ starts) {
  const int gi = blockIdx.x * 256 + threadIdx.x;   // over B*D
  const int d  = gi % D_;
  const int b  = gi / D_;
  float x = -INFINITY;                              // "no prior state"
  for (int c = 0; c < C_; ++c) {
    const int idx = (b * C_ + c) * D_ + d;
    starts[idx] = x;
    x = logaddexp_f(sum_lc[idx] + x, sum_lv[idx]);
  }
}

__global__ __launch_bounds__(256) void scan_stage3_k(
    const float* __restrict__ lcb, const float* __restrict__ lvb,
    const float* __restrict__ starts, const float* __restrict__ h0f,
    float* __restrict__ h1f, bf16_t* __restrict__ h1b,
    float* __restrict__ nh) {
  const int gi = blockIdx.x * 256 + threadIdx.x;   // over B*C*D
  const int d  = gi % D_;
  const int bc = gi / D_;
  const int c  = bc % C_;
  const int b  = bc / C_;

  float logh = starts[gi];
  for (int j = 0; j < L_; ++j) {
    const int s    = c * L_ + j;
    const size_t r = (size_t)(b * S_ + s);
    logh = logaddexp_f(lcb[r * D_ + d] + logh, lvb[r * D_ + d]);
    const float gru = expf(logh);
    const float h1  = gru + h0f[r * D_ + d];
    h1f[r * D_ + d] = h1;
    h1b[r * D_ + d] = (bf16_t)h1;
    if (s == S_ - 1) nh[b * D_ + d] = gru;
  }
}

// ---------------------------------------------------------------------------
// Loss: online logsumexp per row over V, gather label logit, atomic mean.
// ---------------------------------------------------------------------------
__global__ void zero_loss_k(float* out) { out[0] = 0.0f; }

__global__ __launch_bounds__(256) void loss_k(const float* __restrict__ logits,
                                              const int* __restrict__ x,
                                              float* __restrict__ out) {
  const int m = blockIdx.x;
  const int b = m / S_, s = m - b * S_;
  const float* row = logits + (size_t)m * V_;
  const int tid = threadIdx.x;

  float mx = -3.0e38f, sm = 0.0f;
  for (int i = tid; i < V_; i += 256) {
    const float v  = row[i];
    const float nm = fmaxf(mx, v);
    sm = sm * expf(mx - nm) + expf(v - nm);
    mx = nm;
  }
  __shared__ float smx[256], ssm[256];
  smx[tid] = mx; ssm[tid] = sm; __syncthreads();
  for (int o = 128; o > 0; o >>= 1) {
    if (tid < o) {
      const float m2 = smx[tid + o], s2 = ssm[tid + o];
      const float M  = fmaxf(smx[tid], m2);
      ssm[tid] = ssm[tid] * expf(smx[tid] - M) + s2 * expf(m2 - M);
      smx[tid] = M;
    }
    __syncthreads();
  }
  if (tid == 0) {
    const float lse   = smx[0] + logf(ssm[0]);
    const int   label = x[b * S1_ + s + 1];
    atomicAdd(out, (lse - row[label]) * (1.0f / (float)(B_ * S_)));
  }
}

// ---------------------------------------------------------------------------
// Launcher
// ---------------------------------------------------------------------------
extern "C" void kernel_launch(void* const* d_in, const int* in_sizes, int n_in,
                              void* d_out, int out_size, void* d_ws,
                              size_t ws_size, hipStream_t stream) {
  (void)in_sizes; (void)n_in; (void)out_size; (void)ws_size;

  const int*   x      = (const int*)  d_in[0];
  const float* emb    = (const float*)d_in[1];
  const float* gamma1 = (const float*)d_in[2];
  const float* W_hg   = (const float*)d_in[3];
  const float* W1     = (const float*)d_in[4];
  const float* b1     = (const float*)d_in[5];
  const float* W2     = (const float*)d_in[6];
  const float* b2     = (const float*)d_in[7];
  const float* gamma2 = (const float*)d_in[8];
  const float* Wlog   = (const float*)d_in[9];

  float* out    = (float*)d_out;
  float* logits = out + 1;
  float* nh     = out + 1 + (size_t)B_ * S_ * V_;

  // Workspace carve-up (all regions written each call before being read)
  char* p = (char*)d_ws;
  auto carve = [&](size_t bytes) {
    char* r = p;
    p += (bytes + 255) & ~(size_t)255;
    return r;
  };
  bf16_t* Whgb  = (bf16_t*)carve((size_t)D_ * 2 * D_ * 2);
  bf16_t* W1b   = (bf16_t*)carve((size_t)D_ * 4 * D_ * 2);
  bf16_t* W2b   = (bf16_t*)carve((size_t)4 * D_ * D_ * 2);
  bf16_t* Wlb   = (bf16_t*)carve((size_t)D_ * VP_ * 2);
  float*  h0f   = (float*) carve((size_t)T_ * D_ * 4);
  bf16_t* h0b   = (bf16_t*)carve((size_t)T_ * D_ * 2);
  float*  hg    = (float*) carve((size_t)T_ * 2 * D_ * 4);
  float*  h1f   = (float*) carve((size_t)T_ * D_ * 4);
  bf16_t* h1b   = (bf16_t*)carve((size_t)T_ * D_ * 2);
  bf16_t* actb  = (bf16_t*)carve((size_t)T_ * 4 * D_ * 2);
  float*  h2f   = (float*) carve((size_t)T_ * D_ * 4);
  bf16_t* xb    = (bf16_t*)carve((size_t)T_ * D_ * 2);
  float*  lcb   = (float*) carve((size_t)T_ * D_ * 4);
  float*  lvb   = (float*) carve((size_t)T_ * D_ * 4);
  float*  sumlc = (float*) carve((size_t)B_ * C_ * D_ * 4);
  float*  sumlv = (float*) carve((size_t)B_ * C_ * D_ * 4);
  float*  strts = (float*) carve((size_t)B_ * C_ * D_ * 4);

  // 1) Weight converts (fp32 -> bf16; W_logits padded to VP columns)
  cvt_bf16_k<<<(D_ * 2 * D_ / 4 + 255) / 256, 256, 0, stream>>>(W_hg, Whgb, D_ * 2 * D_);
  cvt_bf16_k<<<(D_ * 4 * D_ / 4 + 255) / 256, 256, 0, stream>>>(W1, W1b, D_ * 4 * D_);
  cvt_bf16_k<<<(4 * D_ * D_ / 4 + 255) / 256, 256, 0, stream>>>(W2, W2b, 4 * D_ * D_);
  cvt_wl_pad_k<<<(D_ * VP_ + 255) / 256, 256, 0, stream>>>(Wlog, Wlb);

  // 2) Embedding + RMSNorm1
  embed_rmsnorm_k<<<T_, 256, 0, stream>>>(x, emb, gamma1, h0f, h0b);

  // 3) hg = h0 @ W_hg   (4096 x 2048 x 1024)
  gemm_wmma_bf16<EP_PLAIN><<<dim3(2 * D_ / BN, T_ / BM), 256, 0, stream>>>(
      h0b, Whgb, hg, nullptr, nullptr, nullptr,
      T_, 2 * D_, D_, 2 * D_, 2 * D_, 2 * D_);

  // 4) minGRU chunked scan + residual + next_hidden
  scan_stage1_k<<<(B_ * C_ * D_) / 256, 256, 0, stream>>>(hg, lcb, lvb, sumlc, sumlv);
  scan_stage2_k<<<(B_ * D_) / 256, 256, 0, stream>>>(sumlc, sumlv, strts);
  scan_stage3_k<<<(B_ * C_ * D_) / 256, 256, 0, stream>>>(lcb, lvb, strts, h0f,
                                                          h1f, h1b, nh);

  // 5) MLP-1: gelu(h1 @ W1 + b1) -> bf16   (4096 x 4096 x 1024)
  gemm_wmma_bf16<EP_GELU><<<dim3(4 * D_ / BN, T_ / BM), 256, 0, stream>>>(
      h1b, W1b, nullptr, actb, b1, nullptr,
      T_, 4 * D_, D_, 4 * D_, 4 * D_, 4 * D_);

  // 6) MLP-2: act @ W2 + b2 + h1 -> fp32   (4096 x 1024 x 4096)
  gemm_wmma_bf16<EP_RES><<<dim3(D_ / BN, T_ / BM), 256, 0, stream>>>(
      actb, W2b, h2f, nullptr, b2, h1f,
      T_, D_, 4 * D_, D_, D_, D_);

  // 7) RMSNorm2 -> bf16
  rmsnorm_bf16_k<<<T_, 256, 0, stream>>>(h2f, gamma2, xb);

  // 8) Logits GEMM straight into d_out (4096 x 50304pad x 1024, store-guard V)
  zero_loss_k<<<1, 1, 0, stream>>>(out);
  gemm_wmma_bf16<EP_LOGITS><<<dim3(VP_ / BN, T_ / BM), 256, 0, stream>>>(
      xb, Wlb, logits, nullptr, nullptr, nullptr,
      T_, VP_, D_, VP_, V_, V_);

  // 9) log-softmax loss
  loss_k<<<T_, 256, 0, stream>>>(logits, x, out);
}
